// PairformerLayer_16793322127894
// MI455X (gfx1250) — matrix operand order repaired and checked by reading, not compiled, not run on hardware
//
// Pairformer layer for MI455X (gfx1250), wave32 + WMMA f32_16x16x32_f16.
// LDS-tiled generic-stride batched GEMM: 128x64 block tile, 8 waves (full WGP),
// each wave 32x32 (4 WMMAs/K-step). Interior tiles take a branch-free
// vectorized staging path (global b128 -> v_cvt_pk_rtz_f16_f32 -> LDS);
// edge tiles use a clamped/select path. A operands via aligned ds_load_b128; B via 16-bit column reads.
// Workspace requirement: 8 regions * 256*256*128 floats = 256 MB.

#include <hip/hip_runtime.h>
#include <hip/hip_bf16.h>
#include <cmath>

typedef __attribute__((ext_vector_type(16))) _Float16 v16h;
typedef __attribute__((ext_vector_type(8)))  _Float16 v8h;
typedef __attribute__((ext_vector_type(2)))  _Float16 h2;
typedef __attribute__((ext_vector_type(8)))  float    v8f;

#define DEVINL __device__ __forceinline__

DEVINL float sigmoidf_(float x) { return 1.f / (1.f + __expf(-x)); }

// ---------------------------------------------------------------------------
// Generic batched GEMM:  C[b] = alpha * A[b] * B[b] (+ bias[n]) (+= C)
//   A[m,k] = A[b*sAb + m*sAm + k*sAk]
//   B[k,n] = B[b*sBb + k*sBk + n*sBn]
//   C[m,n] = C[b*sCb + m*sCm + n*sCn]
// Block tile 128x64, K slab 32. 8 waves arranged 4(m) x 2(n), wave tile 32x32.
// LDS pitches give 16B-aligned ds_load_b128 and conflict-free row strides:
//   A: [128][40] halves (80B rows), B: [32][72] halves (144B rows).
// ---------------------------------------------------------------------------
#define APITCH 40
#define BPITCH 72

__global__ __launch_bounds__(256)
void wgemm_kernel(float* __restrict__ C, const float* __restrict__ A,
                  const float* __restrict__ B,
                  int M, int N, int K,
                  long sAm, long sAk, long sAb,
                  long sBk, long sBn, long sBb,
                  long sCm, long sCn, long sCb,
                  float alpha, const float* __restrict__ bias, int accum) {
  __shared__ _Float16 As[128 * APITCH];  // 10240 B
  __shared__ _Float16 Bs[32 * BPITCH];   //  4608 B

  const int tid  = threadIdx.x;
  const int w    = tid >> 5;
  const int lane = tid & 31;
  const int m0   = blockIdx.y * 128;
  const int n0   = blockIdx.x * 64;
  const long b   = blockIdx.z;
  const float* Ab = A + b * sAb;
  const float* Bb = B + b * sBb;
  float*       Cb = C + b * sCb;

  const int wmb = (w >> 1) * 32;         // wave m offset in block tile (0..96)
  const int wnb = (w & 1) * 32;          // wave n offset in block tile (0/32)

  // staging coordinates: A tile 128x32 (half row/thread, 256 threads),
  //                      B tile 32x64 (8 elements/thread)
  const int arow = tid >> 1, acb = (tid & 1) << 4;
  const int brow = tid >> 3, bcb = (tid & 7) << 3;
  _Float16* aDst = As + arow * APITCH + acb;
  _Float16* bDst = Bs + brow * BPITCH + bcb;
  const float* aRow = Ab + (long)(m0 + arow) * sAm;   // deref only in-bounds

  // operand-read coordinates
  const int mr = lane & 15;
  const int kb = (lane >> 4) << 3;       // 0 or 8 (split-K lane pattern)

  const bool fullMN = (m0 + 128 <= M) && (n0 + 64 <= N);

  v8f acc[2][2] = {{{}, {}}, {{}, {}}};

  auto stA = [&](int j, float f0, float f1) {
    h2 pk;
    { auto t = __builtin_amdgcn_cvt_pkrtz(f0, f1); pk[0] = t[0]; pk[1] = t[1]; }
    *(h2*)(aDst + j) = pk;
  };
  auto stB = [&](int j, float f0, float f1) {
    h2 pk;
    { auto t = __builtin_amdgcn_cvt_pkrtz(f0, f1); pk[0] = t[0]; pk[1] = t[1]; }
    *(h2*)(bDst + j) = pk;
  };

  // branch-free staging for interior tiles (full M/N/K-step)
  auto stage_fast = [&](int k0) {
    if (sAk == 1) {
      const float4* src = (const float4*)(aRow + k0 + acb);
      float4 x0 = src[0], x1 = src[1], x2 = src[2], x3 = src[3];
      stA(0, x0.x, x0.y);  stA(2, x0.z, x0.w);
      stA(4, x1.x, x1.y);  stA(6, x1.z, x1.w);
      stA(8, x2.x, x2.y);  stA(10, x2.z, x2.w);
      stA(12, x3.x, x3.y); stA(14, x3.z, x3.w);
    } else {
      const float* p = aRow + (long)(k0 + acb) * sAk;
      float f[16];
#pragma unroll
      for (int j = 0; j < 16; ++j) { f[j] = *p; p += sAk; }
#pragma unroll
      for (int j = 0; j < 16; j += 2) stA(j, f[j], f[j + 1]);
    }
    const float* bRow = Bb + (long)(k0 + brow) * sBk;
    if (sBn == 1) {
      const float4* src = (const float4*)(bRow + n0 + bcb);
      float4 x0 = src[0], x1 = src[1];
      stB(0, x0.x, x0.y);  stB(2, x0.z, x0.w);
      stB(4, x1.x, x1.y);  stB(6, x1.z, x1.w);
    } else {
      const float* p = bRow + (long)(n0 + bcb) * sBn;
      float f[8];
#pragma unroll
      for (int j = 0; j < 8; ++j) { f[j] = *p; p += sBn; }
#pragma unroll
      for (int j = 0; j < 8; j += 2) stB(j, f[j], f[j + 1]);
    }
  };

  // clamped/select staging for edge tiles (no OOB derefs)
  auto stage_guard = [&](int k0) {
    const int gm = m0 + arow;
    const int gmc = gm < M ? gm : (M - 1);
    const float* ap = Ab + (long)gmc * sAm;
#pragma unroll
    for (int j = 0; j < 16; j += 2) {
      int k1 = k0 + acb + j, k2 = k1 + 1;
      int k1c = k1 < K ? k1 : (K - 1);
      int k2c = k2 < K ? k2 : (K - 1);
      float f0 = ap[(long)k1c * sAk];
      float f1 = ap[(long)k2c * sAk];
      f0 = (gm < M && k1 < K) ? f0 : 0.f;
      f1 = (gm < M && k2 < K) ? f1 : 0.f;
      stA(j, f0, f1);
    }
    const int gk = k0 + brow;
    const int gkc = gk < K ? gk : (K - 1);
    const float* bp = Bb + (long)gkc * sBk;
#pragma unroll
    for (int j = 0; j < 8; j += 2) {
      int n1 = n0 + bcb + j, n2 = n1 + 1;
      int n1c = n1 < N ? n1 : (N - 1);
      int n2c = n2 < N ? n2 : (N - 1);
      float f0 = bp[(long)n1c * sBn];
      float f1 = bp[(long)n2c * sBn];
      f0 = (gk < K && n1 < N) ? f0 : 0.f;
      f1 = (gk < K && n2 < N) ? f1 : 0.f;
      stB(j, f0, f1);
    }
  };

  auto compute = [&]() {
    __syncthreads();
    v16h av[2], bv[2];
#pragma unroll
    for (int ti = 0; ti < 2; ++ti) {
      const _Float16* p = As + (wmb + ti * 16 + mr) * APITCH + kb;
      v8h lo = *(const v8h*)p;
      v8h hi = *(const v8h*)(p + 16);
      av[ti] = __builtin_shufflevector(lo, hi, 0, 1, 2, 3, 4, 5, 6, 7,
                                       8, 9, 10, 11, 12, 13, 14, 15);
    }
#pragma unroll
    for (int tj = 0; tj < 2; ++tj) {
      const _Float16* p = Bs + kb * BPITCH + wnb + tj * 16 + mr;
#pragma unroll
      for (int s = 0; s < 8; ++s) {
        bv[tj][s] = p[s * BPITCH];
        bv[tj][s + 8] = p[(16 + s) * BPITCH];
      }
    }
#pragma unroll
    for (int ti = 0; ti < 2; ++ti)
#pragma unroll
      for (int tj = 0; tj < 2; ++tj)
        acc[ti][tj] = __builtin_amdgcn_wmma_f32_16x16x32_f16(
            false, av[ti], false, bv[tj], (short)0, acc[ti][tj], false, false);
    __syncthreads();
  };

  if (fullMN) {
    const int kFull = K & ~31;
    for (int k0 = 0; k0 < kFull; k0 += 32) { stage_fast(k0); compute(); }
    if (kFull < K) { stage_guard(kFull); compute(); }
  } else {
    for (int k0 = 0; k0 < K; k0 += 32) { stage_guard(k0); compute(); }
  }

  // ---- epilogue: C/D layout lane L -> N=L&15; VGPR r -> M=r+8*(L>=16) ----
  const int nr    = lane & 15;
  const int mbase = (lane >> 4) << 3;
  long colIdx[2];
  float biasv[2] = {0.f, 0.f};
#pragma unroll
  for (int tj = 0; tj < 2; ++tj) {
    int n = n0 + wnb + tj * 16 + nr;
    colIdx[tj] = (long)n * sCn;
    if (bias) biasv[tj] = (fullMN || n < N) ? bias[n < N ? n : 0] : 0.f;
  }
  if (fullMN) {
#pragma unroll
    for (int ti = 0; ti < 2; ++ti) {
      long rIdx = (long)(m0 + wmb + ti * 16 + mbase) * sCm;
#pragma unroll
      for (int r = 0; r < 8; ++r) {
#pragma unroll
        for (int tj = 0; tj < 2; ++tj) {
          long idx = rIdx + colIdx[tj];
          float v = alpha * acc[ti][tj][r] + biasv[tj];
          if (accum) v += Cb[idx];
          Cb[idx] = v;
        }
        rIdx += sCm;
      }
    }
  } else {
#pragma unroll
    for (int ti = 0; ti < 2; ++ti) {
      long rIdx = (long)(m0 + wmb + ti * 16 + mbase) * sCm;
#pragma unroll
      for (int r = 0; r < 8; ++r) {
        int m = m0 + wmb + ti * 16 + mbase + r;
#pragma unroll
        for (int tj = 0; tj < 2; ++tj) {
          int n = n0 + wnb + tj * 16 + nr;
          if (m < M && n < N) {
            long idx = rIdx + colIdx[tj];
            float v = alpha * acc[ti][tj][r] + biasv[tj];
            if (accum) v += Cb[idx];
            Cb[idx] = v;
          }
        }
        rIdx += sCm;
      }
    }
  }
}

// ---------------------------------------------------------------------------
// LayerNorm: one wave per row. tN>0 => gather row r from token (r%tN)*tN+r/tN
// ---------------------------------------------------------------------------
__global__ __launch_bounds__(256)
void ln_kernel(float* __restrict__ out, const float* __restrict__ x,
               const float* __restrict__ w, const float* __restrict__ b,
               int rows, int D, int tN) {
  int wid = blockIdx.x * (blockDim.x >> 5) + (threadIdx.x >> 5);
  if (wid >= rows) return;
  int lane = threadIdx.x & 31;
  long t = wid;
  if (tN > 0) t = (long)(wid % tN) * tN + (wid / tN);
  const float* xr = x + t * (long)D;
  float s = 0.f, ss = 0.f;
  for (int i = lane; i < D; i += 32) { float v = xr[i]; s += v; ss += v * v; }
#pragma unroll
  for (int m = 16; m; m >>= 1) {
    s  += __shfl_xor(s,  m, 32);
    ss += __shfl_xor(ss, m, 32);
  }
  float mu  = s / (float)D;
  float var = ss / (float)D - mu * mu;
  float inv = rsqrtf(var + 1e-5f);
  float* orow = out + (long)wid * D;
  for (int i = lane; i < D; i += 32)
    orow[i] = (xr[i] - mu) * inv * w[i] + b[i];
}

// Softmax over rows of length L, with additive bias row = bias + (r%bias_mod)*L.
__global__ __launch_bounds__(256)
void softmax_bias_kernel(float* __restrict__ x, const float* __restrict__ bias,
                         int rows, int L, int bias_mod) {
  int r = blockIdx.x * (blockDim.x >> 5) + (threadIdx.x >> 5);
  if (r >= rows) return;
  int lane = threadIdx.x & 31;
  float* row = x + (long)r * L;
  const float* br = bias ? bias + (long)(r % bias_mod) * L : nullptr;
  float mx = -3.402823e38f;
  for (int i = lane; i < L; i += 32) {
    float v = row[i] + (br ? br[i] : 0.f);
    row[i] = v;
    mx = fmaxf(mx, v);
  }
#pragma unroll
  for (int m = 16; m; m >>= 1) mx = fmaxf(mx, __shfl_xor(mx, m, 32));
  float s = 0.f;
  for (int i = lane; i < L; i += 32) { float e = __expf(row[i] - mx); row[i] = e; s += e; }
#pragma unroll
  for (int m = 16; m; m >>= 1) s += __shfl_xor(s, m, 32);
  float inv = 1.f / s;
  for (int i = lane; i < L; i += 32) row[i] *= inv;
}

// ---- element-wise kernels -------------------------------------------------
__global__ __launch_bounds__(256)
void gate_mul_kernel(float* __restrict__ o, const float* __restrict__ a,
                     const float* __restrict__ g, long n) {
  long i  = (long)blockIdx.x * blockDim.x + threadIdx.x;
  long st = (long)gridDim.x * blockDim.x;
  for (; i < n; i += st) o[i] = a[i] * sigmoidf_(g[i]);
}
__global__ __launch_bounds__(256)
void gate_res_kernel(float* __restrict__ d, const float* __restrict__ p,
                     const float* __restrict__ g, long n) {
  long i  = (long)blockIdx.x * blockDim.x + threadIdx.x;
  long st = (long)gridDim.x * blockDim.x;
  for (; i < n; i += st) d[i] += p[i] * sigmoidf_(g[i]);
}
__global__ __launch_bounds__(256)
void silu_mul_kernel(float* __restrict__ o, const float* __restrict__ a,
                     const float* __restrict__ b, long n) {
  long i  = (long)blockIdx.x * blockDim.x + threadIdx.x;
  long st = (long)gridDim.x * blockDim.x;
  for (; i < n; i += st) { float x = a[i]; o[i] = x * sigmoidf_(x) * b[i]; }
}

// ---------------------------------------------------------------------------
// Host orchestration
// ---------------------------------------------------------------------------
extern "C" void kernel_launch(void* const* d_in, const int* in_sizes, int n_in,
                              void* d_out, int out_size, void* d_ws, size_t ws_size,
                              hipStream_t stream) {
  (void)in_sizes; (void)n_in; (void)out_size; (void)ws_size;
  const int  NN = 256, DZv = 128, DSv = 384;
  const long NT = (long)NN * NN;       // 65536 tokens
  const long RS = NT * DZv;            // 8388608 floats per ws region (32 MB)

  const float* s_in = (const float*)d_in[0];
  const float* z_in = (const float*)d_in[1];
  auto P = [&](int i) { return (const float*)d_in[i]; };

  float* sO = (float*)d_out;                 // s output [256,384]
  float* zO = sO + (long)NN * DSv;           // z output [256,256,128]

  float* Wb = (float*)d_ws;
  float* R0 = Wb;            // LN outputs
  float* R1 = Wb + 1 * RS;   // t / o / scores(apb)
  float* R2 = Wb + 2 * RS;   // gate / t_ln
  float* R3 = Wb + 3 * RS;   // q / ab(lo) / h1
  float* R4 = Wb + 4 * RS;   // k / ab(hi) / h2
  float* R5 = Wb + 5 * RS;   // v / p_out
  float* R6 = Wb + 6 * RS;   // scores chunk / zbT / g_out
  float* R7 = Wb + 7 * RS;   // tri-att bias [H,N,N] / s_n

  hipMemcpyAsync(sO, s_in, (size_t)NN * DSv * sizeof(float),
                 hipMemcpyDeviceToDevice, stream);
  hipMemcpyAsync(zO, z_in, (size_t)RS * sizeof(float),
                 hipMemcpyDeviceToDevice, stream);

  auto ln = [&](float* out, const float* x, const float* w, const float* b,
                long rows, int D, int tN) {
    dim3 g((unsigned)((rows + 7) / 8)), bl(256);
    ln_kernel<<<g, bl, 0, stream>>>(out, x, w, b, (int)rows, D, tN);
  };
  auto gemm = [&](float* C, const float* A, const float* B, int M, int N, int K,
                  int batch, long sAm, long sAk, long sAb,
                  long sBk, long sBn, long sBb,
                  long sCm, long sCn, long sCb,
                  float alpha, const float* bias, int acc) {
    dim3 g((N + 63) / 64, (M + 127) / 128, batch), bl(256);
    wgemm_kernel<<<g, bl, 0, stream>>>(C, A, B, M, N, K, sAm, sAk, sAb,
                                       sBk, sBn, sBb, sCm, sCn, sCb,
                                       alpha, bias, acc);
  };
  // x @ W.T with W row-major [Nout, Kin]
  auto proj = [&](float* C, const float* X, const float* Wt, int M, int Nout,
                  int Kin, const float* bias, int acc) {
    gemm(C, X, Wt, M, Nout, Kin, 1, Kin, 1, 0, 1, Kin, 0, Nout, 1, 0,
         1.f, bias, acc);
  };
  auto ew = [&](long n) { long b = (n + 255) / 256; return (unsigned)(b > 16384 ? 16384 : b); };
  auto gatemul = [&](float* o, const float* a, const float* g, long n) {
    gate_mul_kernel<<<ew(n), 256, 0, stream>>>(o, a, g, n);
  };
  auto gateres = [&](float* d, const float* p, const float* g, long n) {
    gate_res_kernel<<<ew(n), 256, 0, stream>>>(d, p, g, n);
  };
  auto silumul = [&](float* o, const float* a, const float* b, long n) {
    silu_mul_kernel<<<ew(n), 256, 0, stream>>>(o, a, b, n);
  };
  auto smax = [&](float* x, const float* bias, int rows, int L, int bmod) {
    dim3 g((unsigned)((rows + 7) / 8)), bl(256);
    softmax_bias_kernel<<<g, bl, 0, stream>>>(x, bias, rows, L, bmod);
  };

  // ---------------- triangle multiplicative update ----------------
  auto tri_mul = [&](const float* niw, const float* nib, const float* p_in,
                     const float* g_in, const float* now, const float* nob,
                     const float* p_out, const float* g_out, bool outgoing) {
    ln(R0, zO, niw, nib, NT, DZv, 0);
    proj(R3, R0, p_in, (int)NT, 2 * DZv, DZv, nullptr, 0);   // P -> R3..R4
    proj(R1, R0, g_in, (int)NT, 2 * DZv, DZv, nullptr, 0);   // G -> R1..R2
    gatemul(R3, R3, R1, NT * 2 * DZv);                       // AB = P*sig(G)
    // t[i,j,d] = sum_k a[.,.,d] b[.,.,d] : 128 batched 256x256x256 GEMMs
    if (outgoing)   // bikd,bjkd
      gemm(R1, R3, R3 + DZv, NN, NN, NN, DZv,
           (long)NN * 2 * DZv, 2 * DZv, 1,
           2 * DZv, (long)NN * 2 * DZv, 1,
           (long)NN * DZv, DZv, 1, 1.f, nullptr, 0);
    else            // bkid,bkjd
      gemm(R1, R3, R3 + DZv, NN, NN, NN, DZv,
           2 * DZv, (long)NN * 2 * DZv, 1,
           (long)NN * 2 * DZv, 2 * DZv, 1,
           (long)NN * DZv, DZv, 1, 1.f, nullptr, 0);
    ln(R2, R1, now, nob, NT, DZv, 0);
    proj(R5, R2, p_out, (int)NT, DZv, DZv, nullptr, 0);      // proj(t_ln)
    proj(R6, R0, g_out, (int)NT, DZv, DZv, nullptr, 0);      // gate(x_in)
    gateres(zO, R5, R6, RS);                                 // z += P2*sig(G2)
  };

  // ---------------- triangle attention ----------------
  auto tri_att = [&](const float* lnw, const float* lnb, const float* Wbias,
                     const float* Wq, const float* Wk, const float* Wv,
                     const float* Wg, const float* Wo, bool ending) {
    const int hz = 128;  // HZ*HDZ
    ln(R0, zO, lnw, lnb, NT, DZv, ending ? NN : 0);
    proj(R3, R0, Wq, (int)NT, hz, DZv, nullptr, 0);
    proj(R4, R0, Wk, (int)NT, hz, DZv, nullptr, 0);
    proj(R5, R0, Wv, (int)NT, hz, DZv, nullptr, 0);
    proj(R2, R0, Wg, (int)NT, hz, DZv, nullptr, 0);
    // bias[h,m,n]: store transposed via C strides (sCm=1, sCn=NT)
    gemm(R7, R0, Wbias, (int)NT, 4, DZv, 1,
         DZv, 1, 0, 1, DZv, 0, 1, NT, 0, 1.f, nullptr, 0);
    const float scale = 1.f / sqrtf(32.f);
    for (int h = 0; h < 4; ++h)
      for (int ic = 0; ic < 4; ++ic) {  // 64-row i-chunks keep scores at 16 MB
        long off = (long)ic * 64 * NN * hz;
        // scores[m,n] = q[i,m,h,:]·k[i,n,h,:]
        gemm(R6, R3 + off + h * 32, R4 + off + h * 32, NN, NN, 32, 64,
             hz, 1, (long)NN * hz,
             1, hz, (long)NN * hz,
             NN, 1, (long)NN * NN, scale, nullptr, 0);
        smax(R6, R7 + (long)h * NT, 64 * NN, NN, NN);
        // o[m,d] = sum_n p[m,n] v[i,n,h,d]
        gemm(R1 + off + h * 32, R6, R5 + off + h * 32, NN, 32, NN, 64,
             NN, 1, (long)NN * NN,
             hz, 1, (long)NN * hz,
             hz, 1, (long)NN * hz, 1.f, nullptr, 0);
      }
    gatemul(R1, R1, R2, NT * hz);
    if (!ending)
      proj(zO, R1, Wo, (int)NT, DZv, hz, nullptr, 1);        // z += o @ Wo.T
    else  // write back transposed: z[j',i'] += out[i',j'] (batch over i')
      gemm(zO, R1, Wo, NN, DZv, hz, NN,
           hz, 1, (long)NN * hz,
           1, hz, 0,
           (long)NN * DZv, 1, DZv, 1.f, nullptr, 1);
  };

  // ---------------- transition (SwiGLU-style) ----------------
  auto trans = [&](float* X, long rows, int D, const float* lnw,
                   const float* lnb, const float* fc1, const float* fc2,
                   const float* fc3) {
    int H = 4 * D;
    ln(R0, X, lnw, lnb, rows, D, 0);
    long chunk = RS / H; if (chunk > rows) chunk = rows;
    for (long c0 = 0; c0 < rows; c0 += chunk) {
      int cm = (int)((rows - c0 < chunk) ? (rows - c0) : chunk);
      proj(R3, R0 + c0 * D, fc1, cm, H, D, nullptr, 0);
      proj(R4, R0 + c0 * D, fc2, cm, H, D, nullptr, 0);
      silumul(R3, R3, R4, (long)cm * H);
      proj(X + c0 * D, R3, fc3, cm, D, H, nullptr, 1);       // += residual
    }
  };

  // ================= layer sequence =================
  tri_mul(P(2), P(3), P(4), P(5), P(6), P(7), P(8), P(9), true);
  tri_mul(P(10), P(11), P(12), P(13), P(14), P(15), P(16), P(17), false);
  tri_att(P(18), P(19), P(20), P(21), P(22), P(23), P(24), P(25), false);
  tri_att(P(26), P(27), P(28), P(29), P(30), P(31), P(32), P(33), true);
  trans(zO, NT, DZv, P(34), P(35), P(36), P(37), P(38));

  // ---------------- attention with pair bias (s track) ----------------
  {
    const float *nsw = P(39), *nsb = P(40), *Wq = P(41), *qb = P(42),
                *Wk = P(43), *Wv = P(44), *Wg = P(45), *zlnw = P(46),
                *zlnb = P(47), *Wz = P(48), *Wo = P(49);
    const int HS = 16, HDS = 24;
    ln(R7, sO, nsw, nsb, NN, DSv, 0);                        // s_n
    proj(R3, R7, Wq, NN, DSv, DSv, qb, 0);                   // q (+bias)
    proj(R4, R7, Wk, NN, DSv, DSv, nullptr, 0);
    proj(R5, R7, Wv, NN, DSv, DSv, nullptr, 0);
    proj(R2, R7, Wg, NN, DSv, DSv, nullptr, 0);
    ln(R0, zO, zlnw, zlnb, NT, DZv, 0);
    // zbT[h,i,j] via transposed C store
    gemm(R6, R0, Wz, (int)NT, HS, DZv, 1,
         DZv, 1, 0, 1, DZv, 0, 1, NT, 0, 1.f, nullptr, 0);
    // scores[h,i,j], batch over heads (head offset = h*HDS inside q/k rows)
    gemm(R1, R3, R4, NN, NN, HDS, HS,
         DSv, 1, HDS, 1, DSv, HDS,
         NN, 1, (long)NN * NN, 1.f / sqrtf((float)HDS), nullptr, 0);
    smax(R1, R6, HS * NN, NN, HS * NN);
    // o[i,h*24+d] = sum_j p[h,i,j] v[j,h,d]  -> reuse R4
    gemm(R4, R1, R5, NN, HDS, NN, HS,
         NN, 1, (long)NN * NN,
         DSv, 1, HDS,
         DSv, 1, HDS, 1.f, nullptr, 0);
    gatemul(R4, R4, R2, (long)NN * DSv);
    proj(sO, R4, Wo, NN, DSv, DSv, nullptr, 1);              // s += o @ Wo.T
  }

  trans(sO, NN, DSv, P(50), P(51), P(52), P(53), P(54));
}
